// Seq2Seq_59236188947151
// MI455X (gfx1250) — matrix operand order repaired and checked by reading, not compiled, not run on hardware
//
#include <hip/hip_runtime.h>

// ---------------------------------------------------------------------------
// Seq2Seq (3-layer LSTM enc/dec + attention) for MI455X / gfx1250.
// All recurrent & projection GEMMs run on v_wmma_f32_16x16x32_bf16
// (bf16 inputs, fp32 accumulate) with 2x2 register blocking: each wave
// owns a 32x32 output tile -> 4 WMMAs per K-step off 8 b128 loads
// (2 loads/WMMA, each fragment reused twice). Gate GEMM = 8x64 = 512
// macro-tiles, exactly one per wave (64 blk x 256 thr = 512 waves).
// Single persistent kernel + atomic grid barrier drives the 400-step
// sequential scan; attention/softmax (<4% of FLOPs) run in VALU.
// ---------------------------------------------------------------------------

#define NBLK 64
#define NTHR 256
#define B_   256
#define S_   200
#define T_   200
#define H_   512
#define G_   2048      // 4*H (i,f,g,o gates)
#define E_   128
#define VOC_ 128
#define BH   (B_ * H_)
#define GO_  1

typedef __attribute__((ext_vector_type(16))) __bf16 v16bf;
typedef __attribute__((ext_vector_type(8)))  float  v8f;

struct Params {
  const float* eb[3]; const float* db[3]; const float* ob;
  const int* src; const int* trg;
  const unsigned short* emb;
  const unsigned short* eWih[3]; const unsigned short* eWhh[3];
  const unsigned short* dWih[3]; const unsigned short* dWhh[3];
  const unsigned short* oW;
  unsigned short* mem;           // [B][S][H] bf16 encoder memory
  unsigned short* hst;           // [3][B][H] bf16 hidden state
  float* cst;                    // [3][B][H] fp32 cell state
  float* gates;                  // [B][4H] fp32 gate pre-activations
  unsigned short* inp;           // [B][640] bf16 layer-0 input (emb|ctx)
  float* wbuf;                   // [B][S] attention weights
  float* logits;                 // [B][VOC]
  unsigned int* bar;             // grid barrier counter
  float* out;                    // [B][T] result
};

// ---- bf16 helpers (storage type = u16; RNE convert) -----------------------
__device__ __forceinline__ float b2f(unsigned short u) {
  union { float f; unsigned i; } c; c.i = ((unsigned)u) << 16; return c.f;
}
__device__ __forceinline__ unsigned short f2b(float f) {
  union { float f; unsigned i; } c; c.f = f;
  unsigned r = c.i + 0x7fffu + ((c.i >> 16) & 1u);   // round-to-nearest-even
  return (unsigned short)(r >> 16);
}
__device__ __forceinline__ float sig(float x) { return 1.f / (1.f + __expf(-x)); }

__device__ __forceinline__ float wsum(float v) {
  #pragma unroll
  for (int o = 16; o > 0; o >>= 1) v += __shfl_xor(v, o, 32);
  return v;
}
__device__ __forceinline__ float wmaxr(float v) {
  #pragma unroll
  for (int o = 16; o > 0; o >>= 1) v = fmaxf(v, __shfl_xor(v, o, 32));
  return v;
}

// ---- grid barrier (monotonic counter, zeroed each launch) -----------------
__device__ __forceinline__ void gridbar(unsigned int* bar, unsigned int* gen) {
  __syncthreads();
  if (threadIdx.x == 0) {
    __threadfence();
    __hip_atomic_fetch_add(bar, 1u, __ATOMIC_RELEASE, __HIP_MEMORY_SCOPE_AGENT);
    const unsigned int tgt = (*gen + 1u) * (unsigned)NBLK;
    while (__hip_atomic_load(bar, __ATOMIC_ACQUIRE, __HIP_MEMORY_SCOPE_AGENT) < tgt)
      __builtin_amdgcn_s_sleep(2);
  }
  __syncthreads();
  __threadfence();
  ++(*gen);
}

// ---- WMMA fragment loads --------------------------------------------------
// A 16x32 bf16: lane t -> row m=t%16, K chunks [kh*8,+8) and [16+kh*8,+8)
// B 32x16 bf16: lane t -> col n=t%16, K chunks [kh*8,+8) and [16+kh*8,+8)
union V16u { v16bf v; uint4 u[2]; };

__device__ __forceinline__ v16bf ldA(const unsigned short* __restrict__ row,
                                     int k0, int kh) {
  V16u a;
  const unsigned short* pa = row + k0 + kh * 8;
  a.u[0] = *(const uint4*)(pa);
  a.u[1] = *(const uint4*)(pa + 16);
  return a.v;
}
__device__ __forceinline__ v16bf ldB(const unsigned short* __restrict__ row,
                                     int k0, int kh) {
  V16u b;
  const unsigned short* pb = row + k0 + kh * 8;
  b.u[0] = *(const uint4*)(pb);
  b.u[1] = *(const uint4*)(pb + 16);
  return b.v;
}
#define WMMA_BF16(A, Bf, C) \
  __builtin_amdgcn_wmma_f32_16x16x32_bf16(false, (A), false, (Bf), (short)0, (C), false, false)

// out[B][N] = X[B][Kx] @ Wa[N][Kx]^T (+ Hc[B][512] @ Wh[N][512]^T) + bias
// 2x2 register blocking: one 32x32 output macro-tile per loop iteration.
__device__ void gemm_tiles32(const unsigned short* __restrict__ X, int ldx, int Kx,
                             const unsigned short* __restrict__ Wa,
                             const unsigned short* __restrict__ Hc,
                             const unsigned short* __restrict__ Wh,
                             const float* __restrict__ bias,
                             float* __restrict__ out, int N,
                             int wave, int nwaves, int lane) {
  const int ntn = N >> 5;                 // macro-tiles along N
  const int ntiles = (B_ >> 5) * ntn;     // 8 * ntn
  const int lm = lane & 15;
  const int kh = lane >> 4;
  for (int t = wave; t < ntiles; t += nwaves) {
    const int tm = t / ntn, tn = t - tm * ntn;
    const int m0 = tm * 32 + lm, m1 = m0 + 16;
    const int n0 = tn * 32 + lm, n1 = n0 + 16;
    const unsigned short* xr0 = X + (size_t)m0 * ldx;
    const unsigned short* xr1 = X + (size_t)m1 * ldx;
    const unsigned short* wr0 = Wa + (size_t)n0 * Kx;
    const unsigned short* wr1 = Wa + (size_t)n1 * Kx;
    v8f a00 = {0.f, 0.f, 0.f, 0.f, 0.f, 0.f, 0.f, 0.f};
    v8f a01 = a00, a10 = a00, a11 = a00;
    for (int k0 = 0; k0 < Kx; k0 += 32) {
      const v16bf fa0 = ldA(xr0, k0, kh);
      const v16bf fa1 = ldA(xr1, k0, kh);
      const v16bf fb0 = ldB(wr0, k0, kh);
      const v16bf fb1 = ldB(wr1, k0, kh);
      a00 = WMMA_BF16(fa0, fb0, a00);
      a01 = WMMA_BF16(fa0, fb1, a01);
      a10 = WMMA_BF16(fa1, fb0, a10);
      a11 = WMMA_BF16(fa1, fb1, a11);
    }
    if (Hc) {
      const unsigned short* hr0 = Hc + (size_t)m0 * H_;
      const unsigned short* hr1 = Hc + (size_t)m1 * H_;
      const unsigned short* vr0 = Wh + (size_t)n0 * H_;
      const unsigned short* vr1 = Wh + (size_t)n1 * H_;
      for (int k0 = 0; k0 < H_; k0 += 32) {
        const v16bf fa0 = ldA(hr0, k0, kh);
        const v16bf fa1 = ldA(hr1, k0, kh);
        const v16bf fb0 = ldB(vr0, k0, kh);
        const v16bf fb1 = ldB(vr1, k0, kh);
        a00 = WMMA_BF16(fa0, fb0, a00);
        a01 = WMMA_BF16(fa0, fb1, a01);
        a10 = WMMA_BF16(fa1, fb0, a10);
        a11 = WMMA_BF16(fa1, fb1, a11);
      }
    }
    const float bv0 = bias[n0];
    const float bv1 = bias[n1];
    const int r0 = tm * 32 + kh * 8;       // rows for a00/a01
    const int r1 = r0 + 16;                // rows for a10/a11
    #pragma unroll
    for (int j = 0; j < 8; ++j) {
      out[(size_t)(r0 + j) * N + n0] = a00[j] + bv0;
      out[(size_t)(r0 + j) * N + n1] = a01[j] + bv1;
      out[(size_t)(r1 + j) * N + n0] = a10[j] + bv0;
      out[(size_t)(r1 + j) * N + n1] = a11[j] + bv1;
    }
  }
}

// ---- LSTM elementwise -----------------------------------------------------
__device__ void lstm_ew(const float* __restrict__ g, float* __restrict__ c,
                        unsigned short* __restrict__ h,
                        unsigned short* __restrict__ memslot, int gtid, int nth) {
  for (int i = gtid; i < BH; i += nth) {
    const int b = i >> 9, j = i & (H_ - 1);
    const float* gr = g + (size_t)b * G_;
    const float iv = sig(gr[j]);
    const float fv = sig(gr[H_ + j]);
    const float gv = tanhf(gr[2 * H_ + j]);
    const float ov = sig(gr[3 * H_ + j]);
    const float c2 = fv * c[i] + iv * gv;
    const float h2 = ov * tanhf(c2);
    c[i] = c2;
    const unsigned short hb = f2b(h2);
    h[i] = hb;
    if (memslot) memslot[(size_t)b * S_ * H_ + j] = hb;
  }
}

// ---- fp32 -> bf16 weight conversion ---------------------------------------
__global__ void k_f2b(const float* __restrict__ s, unsigned short* __restrict__ d, int n) {
  const int i = blockIdx.x * blockDim.x + threadIdx.x;
  if (i < n) d[i] = f2b(s[i]);
}

// ---- persistent seq2seq kernel --------------------------------------------
__global__ __launch_bounds__(NTHR) void seq2seq_persistent(Params p) {
  const int tid = threadIdx.x;
  const int gtid = blockIdx.x * NTHR + tid;
  const int nth = NBLK * NTHR;
  const int wave = gtid >> 5;
  const int nwaves = nth >> 5;
  const int lane = tid & 31;
  unsigned int gen = 0;

  // ===================== Encoder: 200 steps =====================
  for (int t = 0; t < S_; ++t) {
    for (int i = gtid; i < B_ * E_; i += nth) {         // embed gather
      const int b = i >> 7, e = i & (E_ - 1);
      const int tok = p.src[b * S_ + t];
      p.inp[b * E_ + e] = p.emb[tok * E_ + e];
    }
    gridbar(p.bar, &gen);
    for (int l = 0; l < 3; ++l) {
      const unsigned short* X = (l == 0) ? p.inp : p.hst + (size_t)(l - 1) * BH;
      const int Kx = (l == 0) ? E_ : H_;
      gemm_tiles32(X, Kx, Kx, p.eWih[l], p.hst + (size_t)l * BH, p.eWhh[l],
                   p.eb[l], p.gates, G_, wave, nwaves, lane);
      gridbar(p.bar, &gen);
      lstm_ew(p.gates, p.cst + (size_t)l * BH, p.hst + (size_t)l * BH,
              (l == 2) ? (p.mem + (size_t)t * H_) : (unsigned short*)nullptr,
              gtid, nth);
      gridbar(p.bar, &gen);
    }
  }

  // ===================== Decoder: 200 steps =====================
  for (int t = 0; t < T_; ++t) {
    const unsigned short* q = p.hst + (size_t)2 * BH;   // previous top hidden
    // attention scores: one wave per (b,s) dot product over H
    for (int d = wave; d < B_ * S_; d += nwaves) {
      const int b = d / S_, s = d - b * S_;
      const unsigned short* mr = p.mem + ((size_t)b * S_ + s) * H_;
      const unsigned short* qr = q + (size_t)b * H_;
      float acc = 0.f;
      for (int h = lane; h < H_; h += 32) acc += b2f(mr[h]) * b2f(qr[h]);
      acc = wsum(acc);
      if (lane == 0) p.wbuf[b * S_ + s] = acc;
    }
    gridbar(p.bar, &gen);
    // softmax over S per batch row (one wave per row)
    for (int b = wave; b < B_; b += nwaves) {
      float* row = p.wbuf + (size_t)b * S_;
      float tv[7]; float mx = -3.0e38f;
      #pragma unroll
      for (int u = 0; u < 7; ++u) {
        const int s = lane + u * 32;
        tv[u] = (s < S_) ? row[s] : -3.0e38f; mx = fmaxf(mx, tv[u]);
      }
      mx = wmaxr(mx);
      float sum = 0.f;
      #pragma unroll
      for (int u = 0; u < 7; ++u) {
        const int s = lane + u * 32;
        tv[u] = (s < S_) ? __expf(tv[u] - mx) : 0.f; sum += tv[u];
      }
      sum = wsum(sum);
      const float inv = 1.f / sum;
      #pragma unroll
      for (int u = 0; u < 7; ++u) {
        const int s = lane + u * 32;
        if (s < S_) row[s] = tv[u] * inv;
      }
    }
    gridbar(p.bar, &gen);
    // context vector + target-embedding into inp = [emb | ctx]
    for (int i = gtid; i < B_ * H_; i += nth) {
      const int b = i >> 9, j = i & (H_ - 1);
      const float* wr = p.wbuf + (size_t)b * S_;
      const unsigned short* mc = p.mem + (size_t)b * S_ * H_ + j;
      float acc = 0.f;
      for (int s = 0; s < S_; ++s) acc += wr[s] * b2f(mc[(size_t)s * H_]);
      p.inp[b * 640 + E_ + j] = f2b(acc);
    }
    for (int i = gtid; i < B_ * E_; i += nth) {
      const int b = i >> 7, e = i & (E_ - 1);
      const int tok = (t == 0) ? GO_ : p.trg[b * T_ + t - 1];
      p.inp[b * 640 + e] = p.emb[tok * E_ + e];
    }
    gridbar(p.bar, &gen);
    // 3 LSTM layers
    for (int l = 0; l < 3; ++l) {
      const unsigned short* X = (l == 0) ? p.inp : p.hst + (size_t)(l - 1) * BH;
      const int Kx = (l == 0) ? (E_ + H_) : H_;
      gemm_tiles32(X, Kx, Kx, p.dWih[l], p.hst + (size_t)l * BH, p.dWhh[l],
                   p.db[l], p.gates, G_, wave, nwaves, lane);
      gridbar(p.bar, &gen);
      lstm_ew(p.gates, p.cst + (size_t)l * BH, p.hst + (size_t)l * BH,
              (unsigned short*)nullptr, gtid, nth);
      gridbar(p.bar, &gen);
    }
    // vocab projection (WMMA) then log-softmax + target gather
    gemm_tiles32(p.hst + (size_t)2 * BH, H_, H_, p.oW, nullptr, nullptr,
                 p.ob, p.logits, VOC_, wave, nwaves, lane);
    gridbar(p.bar, &gen);
    for (int b = wave; b < B_; b += nwaves) {
      const float* row = p.logits + (size_t)b * VOC_;
      float v[4]; float mx = -3.0e38f;
      #pragma unroll
      for (int u = 0; u < 4; ++u) { v[u] = row[lane + u * 32]; mx = fmaxf(mx, v[u]); }
      mx = wmaxr(mx);
      float sum = 0.f;
      #pragma unroll
      for (int u = 0; u < 4; ++u) sum += __expf(v[u] - mx);
      sum = wsum(sum);
      if (lane == 0) {
        const int y = p.trg[b * T_ + t];
        p.out[b * T_ + t] = row[y] - mx - __logf(sum);
      }
    }
    gridbar(p.bar, &gen);
  }
}

// ---------------------------------------------------------------------------
extern "C" void kernel_launch(void* const* d_in, const int* in_sizes, int n_in,
                              void* d_out, int out_size, void* d_ws, size_t ws_size,
                              hipStream_t stream) {
  (void)in_sizes; (void)n_in; (void)out_size; (void)ws_size;
  char* w = (char*)d_ws;
  size_t off = 0;
  auto take = [&](size_t bytes) -> char* {
    char* ptr = w + off;
    off += (bytes + 255) & ~(size_t)255;
    return ptr;
  };

  // --- state that must be zero at launch (one memset covers all) ---
  unsigned int*   bar = (unsigned int*)  take(sizeof(unsigned int));
  unsigned short* hst = (unsigned short*)take((size_t)3 * BH * 2);
  float*          cst = (float*)         take((size_t)3 * BH * 4);
  const size_t zbytes = off;
  // --- scratch (fully written before read each launch) ---
  float*          gates  = (float*)         take((size_t)B_ * G_ * 4);
  unsigned short* inp    = (unsigned short*)take((size_t)B_ * (E_ + H_) * 2);
  float*          wbuf   = (float*)         take((size_t)B_ * S_ * 4);
  float*          logits = (float*)         take((size_t)B_ * VOC_ * 4);
  unsigned short* mem    = (unsigned short*)take((size_t)B_ * S_ * H_ * 2);
  // --- bf16 weights ---
  unsigned short* emb_b = (unsigned short*)take((size_t)VOC_ * E_ * 2);
  unsigned short *eWih[3], *eWhh[3], *dWih[3], *dWhh[3];
  const int eK[3] = {E_, H_, H_};
  const int dK[3] = {E_ + H_, H_, H_};
  for (int l = 0; l < 3; ++l) {
    eWih[l] = (unsigned short*)take((size_t)G_ * eK[l] * 2);
    eWhh[l] = (unsigned short*)take((size_t)G_ * H_ * 2);
  }
  for (int l = 0; l < 3; ++l) {
    dWih[l] = (unsigned short*)take((size_t)G_ * dK[l] * 2);
    dWhh[l] = (unsigned short*)take((size_t)G_ * H_ * 2);
  }
  unsigned short* oW = (unsigned short*)take((size_t)VOC_ * H_ * 2);

  hipMemsetAsync(d_ws, 0, zbytes, stream);

  // fp32 -> bf16 weight conversion (deterministic, every launch)
  struct CV { int idx; unsigned short* d; int n; } cv[] = {
    {0, emb_b, VOC_ * E_},
    {1, eWih[0], G_ * E_},           {2, eWhh[0], G_ * H_},
    {4, eWih[1], G_ * H_},           {5, eWhh[1], G_ * H_},
    {7, eWih[2], G_ * H_},           {8, eWhh[2], G_ * H_},
    {10, dWih[0], G_ * (E_ + H_)},   {11, dWhh[0], G_ * H_},
    {13, dWih[1], G_ * H_},          {14, dWhh[1], G_ * H_},
    {16, dWih[2], G_ * H_},          {17, dWhh[2], G_ * H_},
    {19, oW, VOC_ * H_},
  };
  for (int i = 0; i < (int)(sizeof(cv) / sizeof(cv[0])); ++i)
    k_f2b<<<(cv[i].n + 255) / 256, 256, 0, stream>>>(
        (const float*)d_in[cv[i].idx], cv[i].d, cv[i].n);

  Params p;
  p.eb[0] = (const float*)d_in[3];  p.eb[1] = (const float*)d_in[6];  p.eb[2] = (const float*)d_in[9];
  p.db[0] = (const float*)d_in[12]; p.db[1] = (const float*)d_in[15]; p.db[2] = (const float*)d_in[18];
  p.ob  = (const float*)d_in[20];
  p.src = (const int*)d_in[21];
  p.trg = (const int*)d_in[22];
  p.emb = emb_b;
  for (int l = 0; l < 3; ++l) {
    p.eWih[l] = eWih[l]; p.eWhh[l] = eWhh[l];
    p.dWih[l] = dWih[l]; p.dWhh[l] = dWhh[l];
  }
  p.oW = oW;
  p.mem = mem; p.hst = hst; p.cst = cst;
  p.gates = gates; p.inp = inp; p.wbuf = wbuf; p.logits = logits;
  p.bar = bar; p.out = (float*)d_out;

  seq2seq_persistent<<<NBLK, NTHR, 0, stream>>>(p);
}
